// MoELayer_41592463294485
// MI455X (gfx1250) — compile-verified, matrix operand order checked by reading, not run on hardware
//
#include <hip/hip_runtime.h>
#include <hip/hip_bf16.h>

typedef __attribute__((ext_vector_type(16))) __bf16        v16bf;
typedef __attribute__((ext_vector_type(8)))  float         v8f;
typedef __attribute__((ext_vector_type(4)))  float         v4f;
typedef __attribute__((ext_vector_type(4)))  unsigned int  v4u;

#define NE 8       // experts
#define DD 512     // model dim
#define HH 2048    // hidden dim
#define NT 4096    // tokens (2*2048)
#define MT 32      // tokens per FFN block
#define HC 128     // H chunk per iteration (8 waves * 16 cols)

// tiles per expert: W1: (512/32 K-tiles)x(2048/16 N-tiles)=2048, W2: (2048/32)x(512/16)=2048
#define TPE 2048
#define TILE_U32 256   // 32 lanes * 8 uints = 1KB per 32x16 bf16 fragment tile

// float -> bf16 round-to-nearest-even (inputs are finite)
__device__ __forceinline__ unsigned short f2bf(float f) {
    union { float f; unsigned int u; } c; c.f = f;
    unsigned int u = c.u;
    u += 0x7fffu + ((u >> 16) & 1u);
    return (unsigned short)(u >> 16);
}

// ---------------------------------------------------------------------------
// Pre-convert W1/W2 (fp32) into bf16 WMMA B-fragment tiles, one wave per
// 32(K)x16(N) tile. Tile layout: lane l's q0 (elems 0..7) at tile + l*16B,
// q1 (elems 8..15) at tile + 512B + l*16B -> FFN loads are 2x b128, coalesced.
// Tiles ordered n-major/k-minor so K-loops stream contiguous 1KB tiles.
// ---------------------------------------------------------------------------
__global__ __launch_bounds__(256) void moe_convert(
    const float* __restrict__ W1, const float* __restrict__ W2,
    unsigned int* __restrict__ W1bf, unsigned int* __restrict__ W2bf) {
    const int tid = threadIdx.x;
    const int wave = tid >> 5, lane = tid & 31;
    const int nl = lane & 15, half = lane >> 4, kbase = half * 8;
    const int gt = blockIdx.x * 8 + wave;        // tile id, 0 .. NE*TPE-1

    const float* src; int stride; int k0, n0;
    unsigned int* dst;
    const int e = gt >> 11, r = gt & (TPE - 1);
    if (blockIdx.y == 0) {                        // W1: 128 n-tiles x 16 k-tiles
        int nt = r >> 4, kt = r & 15;
        k0 = kt * 32; n0 = nt * 16;
        src = W1 + (size_t)e * DD * HH; stride = HH;
        dst = W1bf + (size_t)gt * TILE_U32;
    } else {                                      // W2: 32 n-tiles x 64 k-tiles
        int nt = r >> 6, kt = r & 63;
        k0 = kt * 32; n0 = nt * 16;
        src = W2 + (size_t)e * HH * DD; stride = DD;
        dst = W2bf + (size_t)gt * TILE_U32;
    }
    // element j of the fragment: K = k0 + kbase + j + (j>=8 ? 8 : 0), N = n0+nl
    const float* p = src + (size_t)(k0 + kbase) * stride + n0 + nl;
    unsigned int u[8];
#pragma unroll
    for (int t = 0; t < 8; ++t) {
        int j0 = 2 * t;
        int K0 = j0 + ((j0 >= 8) ? 8 : 0);
        float a = p[(size_t)K0 * stride];
        float b = p[(size_t)(K0 + 1) * stride];
        u[t] = (unsigned int)f2bf(a) | ((unsigned int)f2bf(b) << 16);
    }
    v4u q0 = { u[0], u[1], u[2], u[3] };
    v4u q1 = { u[4], u[5], u[6], u[7] };
    *(v4u*)(dst + lane * 4)       = q0;
    *(v4u*)(dst + 128 + lane * 4) = q1;
}

// ---------------------------------------------------------------------------
// Gate: logits = x@Wg + bg, softmax, top-2, build per-expert token lists.
// One wave32 per token, 8 tokens per 256-thread block.
// ---------------------------------------------------------------------------
__global__ __launch_bounds__(256) void moe_gate(
    const float* __restrict__ x, const float* __restrict__ Wg,
    const float* __restrict__ bg, float* __restrict__ gate_out,
    int* __restrict__ cursors, int* __restrict__ tokList,
    float* __restrict__ wList) {
    __shared__ float WgT[NE * DD];           // transposed: [e][d], conflict-free
    const int tid = threadIdx.x;
    for (int i = tid; i < NE * DD; i += 256) {
        int d = i >> 3, e = i & 7;
        WgT[e * DD + d] = Wg[i];
    }
    __syncthreads();

    const int wave = tid >> 5, lane = tid & 31;
    const int token = blockIdx.x * 8 + wave;
    float acc[NE];
#pragma unroll
    for (int e = 0; e < NE; ++e) acc[e] = 0.f;
    const float* xr = x + token * DD;
    for (int j = 0; j < DD / 32; ++j) {
        float xv = xr[lane + 32 * j];
#pragma unroll
        for (int e = 0; e < NE; ++e) acc[e] += xv * WgT[e * DD + lane + 32 * j];
    }
#pragma unroll
    for (int off = 16; off >= 1; off >>= 1)
#pragma unroll
        for (int e = 0; e < NE; ++e) acc[e] += __shfl_xor(acc[e], off, 32);

    if (lane == 0) {
        float l[NE], p[NE];
        float mx = -3.4e38f;
#pragma unroll
        for (int e = 0; e < NE; ++e) { l[e] = acc[e] + bg[e]; mx = fmaxf(mx, l[e]); }
        float s = 0.f;
#pragma unroll
        for (int e = 0; e < NE; ++e) { p[e] = expf(l[e] - mx); s += p[e]; }
        float inv = 1.f / s;
#pragma unroll
        for (int e = 0; e < NE; ++e) { p[e] *= inv; gate_out[token * NE + e] = p[e]; }
        // top-2, ties -> lower index (matches jax.lax.top_k)
        int i1 = 0; float p1 = p[0];
#pragma unroll
        for (int e = 1; e < NE; ++e) if (p[e] > p1) { p1 = p[e]; i1 = e; }
        int i2 = (i1 == 0) ? 1 : 0; float p2 = p[i2];
#pragma unroll
        for (int e = 0; e < NE; ++e) if (e != i1 && p[e] > p2) { p2 = p[e]; i2 = e; }
        int pos = atomicAdd(&cursors[i1], 1);
        tokList[i1 * NT + pos] = token; wList[i1 * NT + pos] = p1;
        pos = atomicAdd(&cursors[i2], 1);
        tokList[i2 * NT + pos] = token; wList[i2 * NT + pos] = p2;
    }
}

// ---------------------------------------------------------------------------
// FFN: per (expert, 32-token tile): y = relu(X@W1+b1)@W2 + b2, scaled by the
// gate weight, scatter-added into out. bf16 WMMA 16x16x32, f32 accumulate.
// B operands come pre-packed from W1bf/W2bf (2x global b128 per WMMA);
// A operands come from LDS (2x ds b128, conflict-free padded strides).
// ---------------------------------------------------------------------------
__global__ __launch_bounds__(256) void moe_ffn(
    const float* __restrict__ x,
    const unsigned int* __restrict__ W1bf, const float* __restrict__ b1,
    const unsigned int* __restrict__ W2bf, const float* __restrict__ b2,
    const int* __restrict__ cursors, const int* __restrict__ tokList,
    const float* __restrict__ wList, float* __restrict__ out) {
    __shared__ __attribute__((aligned(16))) unsigned int   Xs[MT * 260]; // 32 x 512 bf16
    __shared__ __attribute__((aligned(16))) unsigned short Hs[MT * 136]; // 32 x 128 bf16
    __shared__ float wLds[MT];
    __shared__ int   tokLds[MT];

    const int e = blockIdx.y;
    const int count = cursors[e];
    const int tile0 = blockIdx.x * MT;
    if (tile0 >= count) return;            // uniform per block

    const int tid = threadIdx.x;
    if (tid < MT) {
        int idx = tile0 + tid;
        int tok = 0; float wt = 0.f;
        if (idx < count) { tok = tokList[e * NT + idx]; wt = wList[e * NT + idx]; }
        tokLds[tid] = tok; wLds[tid] = wt;
    }
    __syncthreads();

    {   // stage token rows into LDS as packed bf16: 8 threads x 64 floats per row
        int r  = tid >> 3;
        int c0 = (tid & 7) * 64;
        const v4f* xr = (const v4f*)(x + (size_t)tokLds[r] * DD + c0);
        unsigned int* dst = Xs + r * 260 + (c0 >> 1);
#pragma unroll
        for (int j = 0; j < 16; ++j) {
            v4f f = xr[j];
            dst[2 * j]     = (unsigned int)f2bf(f.x) | ((unsigned int)f2bf(f.y) << 16);
            dst[2 * j + 1] = (unsigned int)f2bf(f.z) | ((unsigned int)f2bf(f.w) << 16);
        }
    }
    __syncthreads();

    const int wave = tid >> 5, lane = tid & 31;
    const int nl = lane & 15, half = lane >> 4;

    v8f yacc[8];
#pragma unroll
    for (int t = 0; t < 8; ++t) yacc[t] = (v8f){};

    const v4u* w1f = (const v4u*)W1bf;
    const v4u* w2f = (const v4u*)W2bf;

    for (int ch = 0; ch < HH / HC; ++ch) {
        const int hc0 = ch * HC;
        // tile base ids (n-major, k-minor; 64 v4u per tile)
        const int base1 = e * TPE + hc0 + wave * 16;          // + kt (0..15)
        const int base2 = e * TPE + wave * 256 + ch * 4;      // + ct*64 + kk

        // ---- stage 1: h[32,16] = relu(X @ W1[:, wave cols] + b1) ----
        v8f h0 = {}, h1 = {};
#pragma unroll 4
        for (int kt = 0; kt < 16; ++kt) {
            union { v4u q[2]; v16bf v; } Bu;
            size_t toff = (size_t)(base1 + kt) * 64;
            Bu.q[0] = w1f[toff + lane];
            Bu.q[1] = w1f[toff + 32 + lane];
            union { v4u q[2]; v16bf v; } Au0, Au1;
            const v4u* r0 = (const v4u*)(Xs + nl * 260);
            const v4u* r1 = (const v4u*)(Xs + (nl + 16) * 260);
            int qi = kt * 4 + half;
            Au0.q[0] = r0[qi]; Au0.q[1] = r0[qi + 2];
            Au1.q[0] = r1[qi]; Au1.q[1] = r1[qi + 2];
            h0 = __builtin_amdgcn_wmma_f32_16x16x32_bf16(false, Au0.v, false, Bu.v,
                                                         (short)0, h0, false, false);
            h1 = __builtin_amdgcn_wmma_f32_16x16x32_bf16(false, Au1.v, false, Bu.v,
                                                         (short)0, h1, false, false);
        }
        float b1v = b1[e * HH + hc0 + wave * 16 + nl];
#pragma unroll
        for (int i = 0; i < 8; ++i) {
            h0[i] = fmaxf(h0[i] + b1v, 0.f);
            h1[i] = fmaxf(h1[i] + b1v, 0.f);
        }
        __syncthreads();                   // prior chunk's Hs readers done
        {
            int col = wave * 16 + nl;
#pragma unroll
            for (int i = 0; i < 8; ++i) {
                int m = i + 8 * half;      // C/D layout: VGPR i -> M=i(+8), N=lane%16
                Hs[m * 136 + col]        = f2bf(h0[i]);
                Hs[(m + 16) * 136 + col] = f2bf(h1[i]);
            }
        }
        __syncthreads();

        // prefetch next chunk's stage-1 weight tiles
        if (ch + 1 < HH / HC)
            __builtin_prefetch((const void*)(w1f + (size_t)(base1 + HC) * 64 + lane), 0, 1);

        // ---- stage 2: y[32, wave*64 ..+64) += h[32,128] @ W2[hc0..hc0+128, :] ----
#pragma unroll
        for (int kk = 0; kk < 4; ++kk) {
            union { v4u q[2]; v16bf v; } Au0, Au1;
            const v4u* r0 = (const v4u*)(Hs + nl * 136);
            const v4u* r1 = (const v4u*)(Hs + (nl + 16) * 136);
            int qi = kk * 4 + half;
            Au0.q[0] = r0[qi]; Au0.q[1] = r0[qi + 2];
            Au1.q[0] = r1[qi]; Au1.q[1] = r1[qi + 2];
#pragma unroll
            for (int ct = 0; ct < 4; ++ct) {
                union { v4u q[2]; v16bf v; } Bu;
                size_t toff = (size_t)(base2 + ct * 64 + kk) * 64;
                Bu.q[0] = w2f[toff + lane];
                Bu.q[1] = w2f[toff + 32 + lane];
                yacc[ct]     = __builtin_amdgcn_wmma_f32_16x16x32_bf16(
                    false, Au0.v, false, Bu.v, (short)0, yacc[ct], false, false);
                yacc[4 + ct] = __builtin_amdgcn_wmma_f32_16x16x32_bf16(
                    false, Au1.v, false, Bu.v, (short)0, yacc[4 + ct], false, false);
            }
        }
    }
    // ---- epilogue: bias, gate-weight scale, scatter-add (2 commutative adds
    // per out element across the whole grid -> bitwise deterministic) ----
#pragma unroll
    for (int ct = 0; ct < 4; ++ct) {
        int col = wave * 64 + ct * 16 + nl;
        float b2v = b2[e * DD + col];
#pragma unroll
        for (int rt = 0; rt < 2; ++rt) {
            v8f c = yacc[rt * 4 + ct];
#pragma unroll
            for (int i = 0; i < 8; ++i) {
                int m = rt * 16 + i + 8 * half;
                if (tile0 + m < count) {
                    atomicAdd(out + (size_t)tokLds[m] * DD + col,
                              (c[i] + b2v) * wLds[m]);
                }
            }
        }
    }
}

// ---------------------------------------------------------------------------
extern "C" void kernel_launch(void* const* d_in, const int* in_sizes, int n_in,
                              void* d_out, int out_size, void* d_ws, size_t ws_size,
                              hipStream_t stream) {
    (void)in_sizes; (void)n_in; (void)out_size; (void)ws_size;
    const float* x  = (const float*)d_in[0];
    const float* Wg = (const float*)d_in[1];
    const float* bg = (const float*)d_in[2];
    const float* W1 = (const float*)d_in[3];
    const float* b1 = (const float*)d_in[4];
    const float* W2 = (const float*)d_in[5];
    const float* b2 = (const float*)d_in[6];

    float* out      = (float*)d_out;               // [NT, DD]
    float* gate_out = out + (size_t)NT * DD;       // [NT, NE]

    char* ws = (char*)d_ws;
    int*   cursors = (int*)ws;                                   // 256 B
    int*   tokList = (int*)(ws + 256);                           // NE*NT ints
    float* wList   = (float*)(ws + 256 + (size_t)NE * NT * 4);   // NE*NT floats
    size_t woff = 1024 + 2 * (size_t)NE * NT * 4;                // 1KB-aligned
    unsigned int* W1bf = (unsigned int*)(ws + woff);                       // 16 MB
    unsigned int* W2bf = (unsigned int*)(ws + woff + (size_t)NE * TPE * 1024); // 16 MB

    hipMemsetAsync(cursors, 0, 256, stream);
    hipMemsetAsync(out, 0, (size_t)NT * DD * sizeof(float), stream);

    moe_convert<<<dim3(NE * TPE / 8, 2), 256, 0, stream>>>(W1, W2, W1bf, W2bf);
    moe_gate<<<NT / 8, 256, 0, stream>>>(x, Wg, bg, gate_out, cursors, tokList, wList);

    dim3 grid(NT / MT, NE);   // worst case: one expert takes all tokens
    moe_ffn<<<grid, 256, 0, stream>>>(x, W1bf, b1, W2bf, b2,
                                      cursors, tokList, wList, out);
}
